// PiKVMoE_21947282882900
// MI455X (gfx1250) — matrix operand order repaired and checked by reading, not compiled, not run on hardware
//
#include <hip/hip_runtime.h>

#define NTOK 8192
#define HID  2048
#define NEXP 8
#define ROWS 40   // padded LDS row stride in ushorts (80 B): 16B-aligned, conflict-free
#define KSTEPS (HID / 32)

typedef __attribute__((ext_vector_type(16))) __bf16       v16bf;
typedef __attribute__((ext_vector_type(8)))  float        v8f;
typedef __attribute__((ext_vector_type(4)))  float        f32x4;
typedef __attribute__((ext_vector_type(4)))  unsigned int u32x4;
typedef __attribute__((ext_vector_type(2)))  unsigned int u32x2;

union Frag { v16bf v; u32x4 q[2]; };

__device__ __forceinline__ unsigned short f2bf(float f) {
    unsigned int u = __float_as_uint(f);
    u += 0x7FFFu + ((u >> 16) & 1u);          // round-to-nearest-even to bf16
    return (unsigned short)(u >> 16);
}
__device__ __forceinline__ float bf2f(unsigned short s) {
    return __uint_as_float(((unsigned int)s) << 16);
}

// ---------------------------------------------------------------------------
// Kernel 1: gate[n, e] = softmax_e( x[n,:] @ Wg[:,e] + bg[e] )
// ---------------------------------------------------------------------------
__global__ __launch_bounds__(256) void gate_kernel(const float* __restrict__ x,
                                                   const float* __restrict__ Wg,
                                                   const float* __restrict__ bg,
                                                   float* __restrict__ gate) {
    const int lane = threadIdx.x & 31;
    const int wave = threadIdx.x >> 5;
    const int n    = blockIdx.x * 8 + wave;

    const float* xr = x + (size_t)n * HID;
    float acc[NEXP];
#pragma unroll
    for (int j = 0; j < NEXP; ++j) acc[j] = 0.0f;

    for (int h = lane; h < HID; h += 32) {
        float xv = xr[h];
        f32x4 w0 = *(const f32x4*)(Wg + (size_t)h * NEXP);
        f32x4 w1 = *(const f32x4*)(Wg + (size_t)h * NEXP + 4);
        acc[0] += xv * w0.x;  acc[1] += xv * w0.y;
        acc[2] += xv * w0.z;  acc[3] += xv * w0.w;
        acc[4] += xv * w1.x;  acc[5] += xv * w1.y;
        acc[6] += xv * w1.z;  acc[7] += xv * w1.w;
    }
#pragma unroll
    for (int off = 16; off > 0; off >>= 1)
#pragma unroll
        for (int j = 0; j < NEXP; ++j)
            acc[j] += __shfl_xor(acc[j], off, 32);

    float l[NEXP];
#pragma unroll
    for (int j = 0; j < NEXP; ++j) l[j] = acc[j] + bg[j];
    float mx = l[0];
#pragma unroll
    for (int j = 1; j < NEXP; ++j) mx = fmaxf(mx, l[j]);
    float s = 0.0f;
#pragma unroll
    for (int j = 0; j < NEXP; ++j) { l[j] = __expf(l[j] - mx); s += l[j]; }
    float inv = 1.0f / s;

    if (lane == 0) {
        f32x4 g0, g1;
        g0.x = l[0]*inv; g0.y = l[1]*inv; g0.z = l[2]*inv; g0.w = l[3]*inv;
        g1.x = l[4]*inv; g1.y = l[5]*inv; g1.z = l[6]*inv; g1.w = l[7]*inv;
        *(f32x4*)(gate + (size_t)n * NEXP)     = g0;
        *(f32x4*)(gate + (size_t)n * NEXP + 4) = g1;
    }
}

// ---------------------------------------------------------------------------
// Pre-pass A: x (fp32, row-major [n][k]) -> xHi/xLo bf16, same layout.
// ---------------------------------------------------------------------------
__global__ __launch_bounds__(256) void convert_x_kernel(const float* __restrict__ x,
                                                        unsigned short* __restrict__ xHi,
                                                        unsigned short* __restrict__ xLo) {
    size_t i = ((size_t)blockIdx.x * 256 + threadIdx.x) * 4;
    f32x4 v = *(const f32x4*)(x + i);
    float vv[4] = {v.x, v.y, v.z, v.w};
    unsigned short hs[4], ls[4];
#pragma unroll
    for (int j = 0; j < 4; ++j) {
        hs[j] = f2bf(vv[j]);
        ls[j] = f2bf(vv[j] - bf2f(hs[j]));
    }
    u32x2 ph, pl;
    ph.x = (unsigned)hs[0] | ((unsigned)hs[1] << 16);
    ph.y = (unsigned)hs[2] | ((unsigned)hs[3] << 16);
    pl.x = (unsigned)ls[0] | ((unsigned)ls[1] << 16);
    pl.y = (unsigned)ls[2] | ((unsigned)ls[3] << 16);
    *(u32x2*)(xHi + i) = ph;
    *(u32x2*)(xLo + i) = pl;
}

// ---------------------------------------------------------------------------
// Pre-pass B: W[e][k][f] (fp32) -> wHiT/wLoT bf16 TRANSPOSED [e][f][k].
// 64x64 tiles through LDS; coalesced read along f, coalesced write along k.
// ---------------------------------------------------------------------------
#define TROWS 72  // 64 k + pad, multiple of 8 ushorts for b128 alignment
__global__ __launch_bounds__(256) void convert_w_kernel(const float* __restrict__ W,
                                                        unsigned short* __restrict__ wHiT,
                                                        unsigned short* __restrict__ wLoT) {
    __shared__ unsigned short tHi[64 * TROWS];
    __shared__ unsigned short tLo[64 * TROWS];
    const int tid = threadIdx.x;
    const int tk  = blockIdx.x * 64;
    const int tf  = blockIdx.y * 64;
    const int e   = blockIdx.z;
    const size_t eoff = (size_t)e * HID * HID;

#pragma unroll
    for (int i = 0; i < 4; ++i) {
        int q  = tid + 256 * i;          // 0..1023 float4 chunks of the 64x64 tile
        int kk = q >> 4;                 // 0..63
        int fq = (q & 15) << 2;          // 0..60
        f32x4 v = *(const f32x4*)(W + eoff + (size_t)(tk + kk) * HID + tf + fq);
        float vv[4] = {v.x, v.y, v.z, v.w};
#pragma unroll
        for (int j = 0; j < 4; ++j) {
            unsigned short h = f2bf(vv[j]);
            unsigned short l = f2bf(vv[j] - bf2f(h));
            tHi[(fq + j) * TROWS + kk] = h;
            tLo[(fq + j) * TROWS + kk] = l;
        }
    }
    __syncthreads();
#pragma unroll
    for (int i = 0; i < 2; ++i) {
        int q   = tid + 256 * i;         // 0..511 16B chunks of transposed tile
        int ff  = q >> 3;                // 0..63
        int seg = (q & 7) << 3;          // 0..56
        u32x4 vh = *(const u32x4*)(&tHi[ff * TROWS + seg]);
        u32x4 vl = *(const u32x4*)(&tLo[ff * TROWS + seg]);
        size_t o = eoff + (size_t)(tf + ff) * HID + tk + seg;
        *(u32x4*)(wHiT + o) = vh;
        *(u32x4*)(wLoT + o) = vl;
    }
}

// ---------------------------------------------------------------------------
// Kernel 2 (fast path): bf16-preconverted fused MoE GEMM.
// 128x128 tile, 8 waves, K-step 32, double-buffered LDS, 1 barrier/step,
// global loads for step s+1 issued under the WMMAs of step s.
// ---------------------------------------------------------------------------
__device__ __forceinline__ void gload(u32x4* st,
                                      const unsigned short* __restrict__ xHi,
                                      const unsigned short* __restrict__ xLo,
                                      const unsigned short* __restrict__ wHiT,
                                      const unsigned short* __restrict__ wLoT,
                                      int m0, int f0, int e, int k0, int tid) {
    const size_t eoff = (size_t)e * HID * HID;
#pragma unroll
    for (int i = 0; i < 2; ++i) {
        int q   = tid + 256 * i;         // 0..511 16B chunks per 8KB tile
        int row = q >> 2;                // 0..127
        int seg = (q & 3) << 3;          // 0,8,16,24 ushorts
        size_t ax = (size_t)(m0 + row) * HID + k0 + seg;
        size_t bx = eoff + (size_t)(f0 + row) * HID + k0 + seg;
        st[i]     = *(const u32x4*)(xHi  + ax);
        st[2 + i] = *(const u32x4*)(xLo  + ax);
        st[4 + i] = *(const u32x4*)(wHiT + bx);
        st[6 + i] = *(const u32x4*)(wLoT + bx);
    }
}

__global__ __launch_bounds__(256) void moe_fast_kernel(const unsigned short* __restrict__ xHi,
                                                       const unsigned short* __restrict__ xLo,
                                                       const unsigned short* __restrict__ wHiT,
                                                       const unsigned short* __restrict__ wLoT,
                                                       const float* __restrict__ b,
                                                       const float* __restrict__ cached,
                                                       const float* __restrict__ gate,
                                                       float* __restrict__ out) {
    __shared__ unsigned short sm[2][4][128 * ROWS];   // [buf][AH,AL,BH,BL]

    const int tid  = threadIdx.x;
    const int lane = tid & 31;
    const int wave = tid >> 5;
    const int wm   = wave >> 2;
    const int wn   = wave & 3;
    const int lrow = lane & 15;
    const int lhi  = lane >> 4;

    const int m0 = blockIdx.x * 128;
    const int f0 = blockIdx.y * 128;

    const v8f vzero = {0.f, 0.f, 0.f, 0.f, 0.f, 0.f, 0.f, 0.f};
    v8f accOut[4][2], accE[4][2];
#pragma unroll
    for (int mi = 0; mi < 4; ++mi)
#pragma unroll
        for (int ni = 0; ni < 2; ++ni) { accOut[mi][ni] = vzero; accE[mi][ni] = vzero; }

    u32x4 st[8];
    gload(st, xHi, xLo, wHiT, wLoT, m0, f0, 0, 0, tid);

    const int TOT = NEXP * KSTEPS;
    for (int s = 0; s < TOT; ++s) {
        const int buf = s & 1;
        // ---- commit tile s to LDS (waits on its global loads) ----
#pragma unroll
        for (int i = 0; i < 2; ++i) {
            int q   = tid + 256 * i;
            int row = q >> 2;
            int seg = (q & 3) << 3;
            int off = row * ROWS + seg;
            *(u32x4*)(&sm[buf][0][off]) = st[i];
            *(u32x4*)(&sm[buf][1][off]) = st[2 + i];
            *(u32x4*)(&sm[buf][2][off]) = st[4 + i];
            *(u32x4*)(&sm[buf][3][off]) = st[6 + i];
        }
        __syncthreads();
        // ---- prefetch tile s+1 (latency hidden behind WMMAs) ----
        if (s + 1 < TOT)
            gload(st, xHi, xLo, wHiT, wLoT, m0, f0, (s + 1) >> 6, ((s + 1) & 63) << 5, tid);

        // ---- fragments (ISA 7.12.2 16-bit layouts) ----
        Frag aH[4], aL[4], bH[2], bL[2];
#pragma unroll
        for (int mi = 0; mi < 4; ++mi) {
            int base = (wm * 64 + mi * 16 + lrow) * ROWS + lhi * 8;
            aH[mi].q[0] = *(const u32x4*)(&sm[buf][0][base]);
            aH[mi].q[1] = *(const u32x4*)(&sm[buf][0][base + 16]);
            aL[mi].q[0] = *(const u32x4*)(&sm[buf][1][base]);
            aL[mi].q[1] = *(const u32x4*)(&sm[buf][1][base + 16]);
        }
#pragma unroll
        for (int ni = 0; ni < 2; ++ni) {
            int base = (wn * 32 + ni * 16 + lrow) * ROWS + lhi * 8;
            bH[ni].q[0] = *(const u32x4*)(&sm[buf][2][base]);
            bH[ni].q[1] = *(const u32x4*)(&sm[buf][2][base + 16]);
            bL[ni].q[0] = *(const u32x4*)(&sm[buf][3][base]);
            bL[ni].q[1] = *(const u32x4*)(&sm[buf][3][base + 16]);
        }
        // ---- 24 WMMAs: split-precision bf16 products ----
#pragma unroll
        for (int mi = 0; mi < 4; ++mi)
#pragma unroll
            for (int ni = 0; ni < 2; ++ni) {
                accE[mi][ni] = __builtin_amdgcn_wmma_f32_16x16x32_bf16(
                    false, aH[mi].v, false, bH[ni].v, (short)0, accE[mi][ni], false, false);
                accE[mi][ni] = __builtin_amdgcn_wmma_f32_16x16x32_bf16(
                    false, aH[mi].v, false, bL[ni].v, (short)0, accE[mi][ni], false, false);
                accE[mi][ni] = __builtin_amdgcn_wmma_f32_16x16x32_bf16(
                    false, aL[mi].v, false, bH[ni].v, (short)0, accE[mi][ni], false, false);
            }

        // ---- end of expert: relu(+bias)+cached, gate-weighted fold ----
        if ((s & (KSTEPS - 1)) == (KSTEPS - 1)) {
            const int e = s >> 6;
            float bv[2], cv[2];
#pragma unroll
            for (int ni = 0; ni < 2; ++ni) {
                int f = f0 + wn * 32 + ni * 16 + lrow;
                bv[ni] = b[(size_t)e * HID + f];
                cv[ni] = cached[(size_t)e * HID + f];
            }
#pragma unroll
            for (int mi = 0; mi < 4; ++mi) {
                int tokBase = m0 + wm * 64 + mi * 16 + lhi * 8;
#pragma unroll
                for (int r = 0; r < 8; ++r) {
                    float g = gate[(size_t)(tokBase + r) * NEXP + e];
#pragma unroll
                    for (int ni = 0; ni < 2; ++ni) {
                        float aV = accE[mi][ni][r] + bv[ni];
                        aV = fmaxf(aV, 0.0f) + cv[ni];
                        accOut[mi][ni][r] += g * aV;
                    }
                }
            }
#pragma unroll
            for (int mi = 0; mi < 4; ++mi)
#pragma unroll
                for (int ni = 0; ni < 2; ++ni) accE[mi][ni] = vzero;
        }
    }

#pragma unroll
    for (int mi = 0; mi < 4; ++mi) {
        int tokBase = m0 + wm * 64 + mi * 16 + lhi * 8;
#pragma unroll
        for (int r = 0; r < 8; ++r)
#pragma unroll
            for (int ni = 0; ni < 2; ++ni) {
                int f = f0 + wn * 32 + ni * 16 + lrow;
                out[(size_t)(tokBase + r) * HID + f] = accOut[mi][ni][r];
            }
    }
}

// ---------------------------------------------------------------------------
// Fallback (small workspace): round-1 convert-in-kernel version.
// ---------------------------------------------------------------------------
__global__ __launch_bounds__(256) void moe_fallback_kernel(const float* __restrict__ x,
                                                           const float* __restrict__ W,
                                                           const float* __restrict__ b,
                                                           const float* __restrict__ cached,
                                                           const float* __restrict__ gate,
                                                           float* __restrict__ out) {
    __shared__ unsigned short AsHi[128 * ROWS];
    __shared__ unsigned short AsLo[128 * ROWS];
    __shared__ unsigned short BsHi[128 * ROWS];
    __shared__ unsigned short BsLo[128 * ROWS];

    const int tid  = threadIdx.x;
    const int lane = tid & 31;
    const int wave = tid >> 5;
    const int wm   = wave >> 2;
    const int wn   = wave & 3;
    const int lrow = lane & 15;
    const int lhi  = lane >> 4;
    const int m0 = blockIdx.x * 128;
    const int f0 = blockIdx.y * 128;

    const v8f vzero = {0.f, 0.f, 0.f, 0.f, 0.f, 0.f, 0.f, 0.f};
    v8f accOut[4][2];
#pragma unroll
    for (int mi = 0; mi < 4; ++mi)
#pragma unroll
        for (int ni = 0; ni < 2; ++ni) accOut[mi][ni] = vzero;

    for (int e = 0; e < NEXP; ++e) {
        v8f accE[4][2];
#pragma unroll
        for (int mi = 0; mi < 4; ++mi)
#pragma unroll
            for (int ni = 0; ni < 2; ++ni) accE[mi][ni] = vzero;

        const float* We = W + (size_t)e * HID * HID;
        for (int k0 = 0; k0 < HID; k0 += 32) {
#pragma unroll
            for (int i = 0; i < 4; ++i) {
                int q = tid + 256 * i, row = q >> 3, kq = (q & 7) << 2;
                f32x4 v = *(const f32x4*)(x + (size_t)(m0 + row) * HID + k0 + kq);
                float vv[4] = {v.x, v.y, v.z, v.w};
                unsigned short hs[4], ls[4];
#pragma unroll
                for (int j = 0; j < 4; ++j) { hs[j] = f2bf(vv[j]); ls[j] = f2bf(vv[j] - bf2f(hs[j])); }
                u32x2 ph, pl;
                ph.x = (unsigned)hs[0] | ((unsigned)hs[1] << 16);
                ph.y = (unsigned)hs[2] | ((unsigned)hs[3] << 16);
                pl.x = (unsigned)ls[0] | ((unsigned)ls[1] << 16);
                pl.y = (unsigned)ls[2] | ((unsigned)ls[3] << 16);
                *(u32x2*)(&AsHi[row * ROWS + kq]) = ph;
                *(u32x2*)(&AsLo[row * ROWS + kq]) = pl;
            }
#pragma unroll
            for (int i = 0; i < 4; ++i) {
                int q = tid + 256 * i, kk = q >> 5, fq = (q & 31) << 2;
                f32x4 v = *(const f32x4*)(We + (size_t)(k0 + kk) * HID + f0 + fq);
                float vv[4] = {v.x, v.y, v.z, v.w};
#pragma unroll
                for (int j = 0; j < 4; ++j) {
                    unsigned short h = f2bf(vv[j]);
                    unsigned short l = f2bf(vv[j] - bf2f(h));
                    BsHi[(fq + j) * ROWS + kk] = h;
                    BsLo[(fq + j) * ROWS + kk] = l;
                }
            }
            __syncthreads();
            Frag aH[4], aL[4], bH[2], bL[2];
#pragma unroll
            for (int mi = 0; mi < 4; ++mi) {
                int base = (wm * 64 + mi * 16 + lrow) * ROWS + lhi * 8;
                aH[mi].q[0] = *(const u32x4*)(&AsHi[base]);
                aH[mi].q[1] = *(const u32x4*)(&AsHi[base + 16]);
                aL[mi].q[0] = *(const u32x4*)(&AsLo[base]);
                aL[mi].q[1] = *(const u32x4*)(&AsLo[base + 16]);
            }
#pragma unroll
            for (int ni = 0; ni < 2; ++ni) {
                int base = (wn * 32 + ni * 16 + lrow) * ROWS + lhi * 8;
                bH[ni].q[0] = *(const u32x4*)(&BsHi[base]);
                bH[ni].q[1] = *(const u32x4*)(&BsHi[base + 16]);
                bL[ni].q[0] = *(const u32x4*)(&BsLo[base]);
                bL[ni].q[1] = *(const u32x4*)(&BsLo[base + 16]);
            }
#pragma unroll
            for (int mi = 0; mi < 4; ++mi)
#pragma unroll
                for (int ni = 0; ni < 2; ++ni) {
                    accE[mi][ni] = __builtin_amdgcn_wmma_f32_16x16x32_bf16(
                        false, aH[mi].v, false, bH[ni].v, (short)0, accE[mi][ni], false, false);
                    accE[mi][ni] = __builtin_amdgcn_wmma_f32_16x16x32_bf16(
                        false, aH[mi].v, false, bL[ni].v, (short)0, accE[mi][ni], false, false);
                    accE[mi][ni] = __builtin_amdgcn_wmma_f32_16x16x32_bf16(
                        false, aL[mi].v, false, bH[ni].v, (short)0, accE[mi][ni], false, false);
                }
            __syncthreads();
        }
        float bv[2], cv[2];
#pragma unroll
        for (int ni = 0; ni < 2; ++ni) {
            int f = f0 + wn * 32 + ni * 16 + lrow;
            bv[ni] = b[(size_t)e * HID + f];
            cv[ni] = cached[(size_t)e * HID + f];
        }
#pragma unroll
        for (int mi = 0; mi < 4; ++mi) {
            int tokBase = m0 + wm * 64 + mi * 16 + lhi * 8;
#pragma unroll
            for (int r = 0; r < 8; ++r) {
                float g = gate[(size_t)(tokBase + r) * NEXP + e];
#pragma unroll
                for (int ni = 0; ni < 2; ++ni) {
                    float aV = accE[mi][ni][r] + bv[ni];
                    aV = fmaxf(aV, 0.0f) + cv[ni];
                    accOut[mi][ni][r] += g * aV;
                }
            }
        }
    }
#pragma unroll
    for (int mi = 0; mi < 4; ++mi) {
        int tokBase = m0 + wm * 64 + mi * 16 + lhi * 8;
#pragma unroll
        for (int r = 0; r < 8; ++r)
#pragma unroll
            for (int ni = 0; ni < 2; ++ni) {
                int f = f0 + wn * 32 + ni * 16 + lrow;
                out[(size_t)(tokBase + r) * HID + f] = accOut[mi][ni][r];
            }
    }
}

// ---------------------------------------------------------------------------
extern "C" void kernel_launch(void* const* d_in, const int* in_sizes, int n_in,
                              void* d_out, int out_size, void* d_ws, size_t ws_size,
                              hipStream_t stream) {
    (void)in_sizes; (void)n_in; (void)out_size;
    const float* x      = (const float*)d_in[0];
    const float* W      = (const float*)d_in[1];
    const float* b      = (const float*)d_in[2];
    const float* Wg     = (const float*)d_in[3];
    const float* bg     = (const float*)d_in[4];
    const float* cached = (const float*)d_in[5];
    float*       out    = (float*)d_out;

    const size_t GATE_B = (size_t)NTOK * NEXP * sizeof(float);          // 256 KB
    const size_t XBF_B  = (size_t)NTOK * HID * sizeof(unsigned short);  // 32 MB each
    const size_t WBF_B  = (size_t)NEXP * HID * HID * sizeof(unsigned short); // 64 MB each
    const size_t NEED   = GATE_B + 2 * XBF_B + 2 * WBF_B;               // ~192.2 MiB

    char* ws = (char*)d_ws;
    float* gateb = (float*)ws;

    gate_kernel<<<NTOK / 8, 256, 0, stream>>>(x, Wg, bg, gateb);

    if (ws_size >= NEED) {
        unsigned short* xHi  = (unsigned short*)(ws + GATE_B);
        unsigned short* xLo  = (unsigned short*)(ws + GATE_B + XBF_B);
        unsigned short* wHiT = (unsigned short*)(ws + GATE_B + 2 * XBF_B);
        unsigned short* wLoT = (unsigned short*)(ws + GATE_B + 2 * XBF_B + WBF_B);

        convert_x_kernel<<<(NTOK * HID) / (4 * 256), 256, 0, stream>>>(x, xHi, xLo);
        convert_w_kernel<<<dim3(HID / 64, HID / 64, NEXP), 256, 0, stream>>>(W, wHiT, wLoT);
        moe_fast_kernel<<<dim3(NTOK / 128, HID / 128), 256, 0, stream>>>(
            xHi, xLo, wHiT, wLoT, b, cached, gateb, out);
    } else {
        moe_fallback_kernel<<<dim3(NTOK / 128, HID / 128), 256, 0, stream>>>(
            x, W, b, cached, gateb, out);
    }
}
